// _PointnetSAModuleBase_53549652247011
// MI455X (gfx1250) — matrix operand order repaired and checked by reading, not compiled, not run on hardware
//
#include <hip/hip_runtime.h>
#include <hip/hip_bf16.h>
#include <cstdint>

// ---------------------------------------------------------------------------
// PointNet++ SA module for MI455X (gfx1250, wave32, WMMA).
//   inputs : xyz(8,4096,3) f32, features(8,4096,64) f32,
//            W1(64,67) b1(64), W2(128,64) b2(128), W3(256,128) b3(256)
//   outputs: new_xyz(8,1024,3) f32 | new_features(8,256,1024) f32 |
//            sample_inds(8,1024) i32   (flat-concatenated in d_out)
// ---------------------------------------------------------------------------

typedef _Float16 half_t;
typedef __attribute__((ext_vector_type(16))) _Float16 v16h;
typedef __attribute__((ext_vector_type(8)))  float    v8f;

#define NB      8
#define NPTS    4096
#define NPOINT  1024
#define NSAMPLE 32
#define R2      0.04f

union V16HU { v16h v; float4 q[2]; };

// wave-scope LDS fence: lanes of one wave exchange data through LDS; the
// compiler cannot see cross-lane deps, so drain the CDNA5 DS counter.
__device__ __forceinline__ void lds_fence() {
    asm volatile("s_wait_dscnt 0" ::: "memory");
}

// A-fragment (16-bit 16x32 MxK layout, ISA 7.12.2): lane L -> row M=L%16,
// halves [0..7] = K base+(L<16?0:8)+0..7, halves [8..15] = same +16.
__device__ __forceinline__ v16h load_a_frag(const half_t* __restrict__ buf,
                                            int stride, int mt, int kt, int lane) {
    int m    = mt * 16 + (lane & 15);
    int koff = kt * 32 + ((lane >= 16) ? 8 : 0);
    const half_t* p = buf + m * stride + koff;
    V16HU u;
    u.q[0] = *(const float4*)(p);
    u.q[1] = *(const float4*)(p + 16);
    return u.v;
}

// B-fragment pre-packed in ws: frag_idx -> 32 lanes x 16 halves contiguous.
__device__ __forceinline__ v16h load_b_frag(const half_t* __restrict__ packed,
                                            int frag_idx, int lane) {
    const half_t* p = packed + ((size_t)frag_idx * 32 + lane) * 16;
    V16HU u;
    u.q[0] = *(const float4*)(p);
    u.q[1] = *(const float4*)(p + 8);
    return u.v;
}

__device__ __forceinline__ v8f wmma_f16(v16h a, v16h b, v8f c) {
    return __builtin_amdgcn_wmma_f32_16x16x32_f16(false, a, false, b,
                                                  (short)0, c, false, false);
}

// ---------------------------------------------------------------------------
// 1) Furthest point sampling: one block per batch.
// ---------------------------------------------------------------------------
__global__ __launch_bounds__(1024) void fps_kernel(const float* __restrict__ xyz,
                                                   float* __restrict__ new_xyz,
                                                   int* __restrict__ sample_inds) {
    __shared__ float sx[NPTS], sy[NPTS], sz[NPTS];
    __shared__ float rwd[32];
    __shared__ int   rwi[32];
    __shared__ int   scur;

    const int b    = blockIdx.x;
    const int tid  = threadIdx.x;
    const int lane = tid & 31;
    const int wid  = tid >> 5;
    const float* P = xyz + (size_t)b * NPTS * 3;

    for (int i = tid; i < NPTS; i += 1024) {
        sx[i] = P[i * 3 + 0];
        sy[i] = P[i * 3 + 1];
        sz[i] = P[i * 3 + 2];
    }
    __syncthreads();

    float dd[4] = {1e10f, 1e10f, 1e10f, 1e10f};
    int cur = 0;

    if (tid == 0) {
        sample_inds[b * NPOINT] = 0;
        float* nx = new_xyz + (size_t)b * NPOINT * 3;
        nx[0] = sx[0]; nx[1] = sy[0]; nx[2] = sz[0];
    }

    for (int it = 1; it < NPOINT; ++it) {
        const float lx = sx[cur], ly = sy[cur], lz = sz[cur];
        float bd = -1.0f; int bi = 0;
        #pragma unroll
        for (int k = 0; k < 4; ++k) {
            const int i = tid + k * 1024;
            const float dx = sx[i] - lx, dy = sy[i] - ly, dz = sz[i] - lz;
            const float d = dx * dx + dy * dy + dz * dz;
            dd[k] = fminf(dd[k], d);
            if (dd[k] > bd) { bd = dd[k]; bi = i; }
        }
        #pragma unroll
        for (int off = 16; off > 0; off >>= 1) {          // wave32 argmax
            const float od = __shfl_xor(bd, off, 32);
            const int   oi = __shfl_xor(bi, off, 32);
            if (od > bd || (od == bd && oi < bi)) { bd = od; bi = oi; }
        }
        if (lane == 0) { rwd[wid] = bd; rwi[wid] = bi; }
        __syncthreads();
        if (wid == 0) {
            float cd = rwd[lane]; int ci = rwi[lane];
            #pragma unroll
            for (int off = 16; off > 0; off >>= 1) {
                const float od = __shfl_xor(cd, off, 32);
                const int   oi = __shfl_xor(ci, off, 32);
                if (od > cd || (od == cd && oi < ci)) { cd = od; ci = oi; }
            }
            if (lane == 0) scur = ci;
        }
        __syncthreads();
        cur = scur;
        if (tid == 0) {
            sample_inds[b * NPOINT + it] = cur;
            float* nx = new_xyz + ((size_t)b * NPOINT + it) * 3;
            nx[0] = sx[cur]; nx[1] = sy[cur]; nx[2] = sz[cur];
        }
        __syncthreads();
    }
}

// ---------------------------------------------------------------------------
// 2) Ball query: one wave32 per center; first NSAMPLE hits in index order.
// ---------------------------------------------------------------------------
__global__ __launch_bounds__(256) void ballq_kernel(const float* __restrict__ xyz,
                                                    const float* __restrict__ new_xyz,
                                                    int* __restrict__ ball_idx) {
    const int lane = threadIdx.x & 31;
    const int w    = threadIdx.x >> 5;
    const int sg   = blockIdx.x * 8 + w;           // 0..8191 global center
    const int b    = sg >> 10;
    const float* c = new_xyz + (size_t)sg * 3;
    const float cx = c[0], cy = c[1], cz = c[2];
    const float* P = xyz + (size_t)b * NPTS * 3;

    int cnt = 0, first = -1;
    for (int base = 0; base < NPTS && cnt < NSAMPLE; base += 32) {
        const int i = base + lane;
        const float dx = P[i * 3 + 0] - cx;
        const float dy = P[i * 3 + 1] - cy;
        const float dz = P[i * 3 + 2] - cz;
        const bool hit = (dx * dx + dy * dy + dz * dz) < R2;
        const unsigned mask = (unsigned)__ballot(hit);
        if (first < 0 && mask) first = base + __ffs((int)mask) - 1;
        const int pre  = __popc(mask & ((1u << lane) - 1u));
        const int slot = cnt + pre;
        if (hit && slot < NSAMPLE)
            ball_idx[(size_t)sg * NSAMPLE + slot] = i;
        cnt += __popc(mask);
    }
    if (cnt > NSAMPLE) cnt = NSAMPLE;
    const int fill = (first < 0) ? 0 : first;
    if (lane >= cnt) ball_idx[(size_t)sg * NSAMPLE + lane] = fill;
}

// ---------------------------------------------------------------------------
// 3) Pack W(N,Kact) f32 into CDNA5 16-bit B-matrix fragments (f16), K padded
//    to KT*32 with zeros. out[t], t = ((kt*NT+nt)*32+lane)*16+e.
// ---------------------------------------------------------------------------
__global__ __launch_bounds__(256) void pack_w(const float* __restrict__ W,
                                              half_t* __restrict__ out,
                                              int Kact, int NT, int total) {
    const int t = blockIdx.x * 256 + threadIdx.x;
    if (t >= total) return;
    const int e    = t & 15;
    const int lane = (t >> 4) & 31;
    const int nt   = (t >> 9) % NT;
    const int kt   = t / (512 * NT);
    const int k = kt * 32 + ((e < 8) ? e : e + 8) + ((lane >= 16) ? 8 : 0);
    const int n = nt * 16 + (lane & 15);
    const float v = (k < Kact) ? W[n * Kact + k] : 0.0f;
    out[t] = (half_t)v;
}

// ---------------------------------------------------------------------------
// 4) Gather + 3-layer MLP (WMMA) + max-pool. One center per wave32.
//    A-fragments are hoisted into registers once per layer (reused over all
//    n-tiles); B-fragments stream from the packed global layout.
// ---------------------------------------------------------------------------
#define WPB 4  // waves per block

__global__ __launch_bounds__(128) void mlp_kernel(
        const float* __restrict__ xyz, const float* __restrict__ feat,
        const float* __restrict__ b1, const float* __restrict__ b2,
        const float* __restrict__ b3,
        const half_t* __restrict__ pW1, const half_t* __restrict__ pW2,
        const half_t* __restrict__ pW3,
        const int* __restrict__ ball_idx, const float* __restrict__ new_xyz,
        float* __restrict__ out_feat) {
    __shared__ __align__(32) half_t ldsA[WPB][32 * 128]; // G(32x96) / X2(32x128)
    __shared__ __align__(32) half_t ldsB[WPB][32 * 64];  // X1(32x64)

    const int lane = threadIdx.x & 31;
    const int w    = threadIdx.x >> 5;
    const int sg   = blockIdx.x * WPB + w;   // global center 0..8191
    const int b    = sg >> 10;
    const int s    = sg & 1023;

    half_t* A = ldsA[w];
    half_t* B = ldsB[w];

    __builtin_prefetch(pW3, 0, 1);           // warm far weights (global_prefetch_b8)

    // ---- gather: lane j builds row j of G = [rel_xyz | features | 0-pad] ----
    {
        const int pj = ball_idx[(size_t)sg * NSAMPLE + lane];
        const float* c3 = new_xyz + (size_t)sg * 3;
        const float* p3 = xyz + ((size_t)b * NPTS + pj) * 3;
        half_t* row = A + lane * 96;
        row[0] = (half_t)(p3[0] - c3[0]);
        row[1] = (half_t)(p3[1] - c3[1]);
        row[2] = (half_t)(p3[2] - c3[2]);
        const float4* f4 = (const float4*)(feat + ((size_t)b * NPTS + pj) * 64);
        #pragma unroll
        for (int q = 0; q < 16; ++q) {
            const float4 v = f4[q];
            row[3 + 4 * q + 0] = (half_t)v.x;
            row[3 + 4 * q + 1] = (half_t)v.y;
            row[3 + 4 * q + 2] = (half_t)v.z;
            row[3 + 4 * q + 3] = (half_t)v.w;
        }
        #pragma unroll
        for (int k = 67; k < 96; ++k) row[k] = (half_t)0.0f;
    }
    lds_fence();

    // ---- layer 1: (32x96) x (96x64) -> relu -> B (stride 64) ----
    {
        v16h a1[2][3];
        #pragma unroll
        for (int mt = 0; mt < 2; ++mt)
            #pragma unroll
            for (int kt = 0; kt < 3; ++kt)
                a1[mt][kt] = load_a_frag(A, 96, mt, kt, lane);

        #pragma unroll
        for (int nt = 0; nt < 4; ++nt) {
            const float bias = b1[nt * 16 + (lane & 15)];
            #pragma unroll
            for (int mt = 0; mt < 2; ++mt) {
                v8f acc = {};
                #pragma unroll
                for (int kt = 0; kt < 3; ++kt)
                    acc = wmma_f16(a1[mt][kt], load_b_frag(pW1, kt * 4 + nt, lane), acc);
                #pragma unroll
                for (int r = 0; r < 8; ++r) {
                    const int m = mt * 16 + r + ((lane >= 16) ? 8 : 0);
                    const int n = nt * 16 + (lane & 15);
                    float v = acc[r] + bias;
                    B[m * 64 + n] = (half_t)(v > 0.0f ? v : 0.0f);
                }
            }
        }
    }
    lds_fence();

    // ---- layer 2: (32x64) x (64x128) -> relu -> A (stride 128) ----
    {
        v16h a2[2][2];
        #pragma unroll
        for (int mt = 0; mt < 2; ++mt)
            #pragma unroll
            for (int kt = 0; kt < 2; ++kt)
                a2[mt][kt] = load_a_frag(B, 64, mt, kt, lane);

        #pragma unroll
        for (int nt = 0; nt < 8; ++nt) {
            const float bias = b2[nt * 16 + (lane & 15)];
            #pragma unroll
            for (int mt = 0; mt < 2; ++mt) {
                v8f acc = {};
                #pragma unroll
                for (int kt = 0; kt < 2; ++kt)
                    acc = wmma_f16(a2[mt][kt], load_b_frag(pW2, kt * 8 + nt, lane), acc);
                #pragma unroll
                for (int r = 0; r < 8; ++r) {
                    const int m = mt * 16 + r + ((lane >= 16) ? 8 : 0);
                    const int n = nt * 16 + (lane & 15);
                    float v = acc[r] + bias;
                    A[m * 128 + n] = (half_t)(v > 0.0f ? v : 0.0f);
                }
            }
        }
    }
    lds_fence();

    // ---- layer 3: (32x128) x (128x256) -> relu -> max over 32 rows ----
    {
        v16h a3[2][4];
        #pragma unroll
        for (int mt = 0; mt < 2; ++mt)
            #pragma unroll
            for (int kt = 0; kt < 4; ++kt)
                a3[mt][kt] = load_a_frag(A, 128, mt, kt, lane);

        #pragma unroll 4
        for (int nt = 0; nt < 16; ++nt) {
            const float bias = b3[nt * 16 + (lane & 15)];
            float cmax = -1e30f;
            #pragma unroll
            for (int mt = 0; mt < 2; ++mt) {
                v8f acc = {};
                #pragma unroll
                for (int kt = 0; kt < 4; ++kt)
                    acc = wmma_f16(a3[mt][kt], load_b_frag(pW3, kt * 16 + nt, lane), acc);
                #pragma unroll
                for (int r = 0; r < 8; ++r) {
                    float v = acc[r] + bias;
                    v = v > 0.0f ? v : 0.0f;
                    cmax = fmaxf(cmax, v);
                }
            }
            cmax = fmaxf(cmax, __shfl_xor(cmax, 16, 32));   // merge row halves
            if (lane < 16) {                                 // (B, 256, S) layout
                const int n = nt * 16 + lane;
                out_feat[((size_t)b * 256 + n) * NPOINT + s] = cmax;
            }
        }
    }
}

// ---------------------------------------------------------------------------
extern "C" void kernel_launch(void* const* d_in, const int* in_sizes, int n_in,
                              void* d_out, int out_size, void* d_ws, size_t ws_size,
                              hipStream_t stream) {
    const float* xyz  = (const float*)d_in[0];
    const float* feat = (const float*)d_in[1];
    const float* W1   = (const float*)d_in[2];
    const float* b1   = (const float*)d_in[3];
    const float* W2   = (const float*)d_in[4];
    const float* b2   = (const float*)d_in[5];
    const float* W3   = (const float*)d_in[6];
    const float* b3   = (const float*)d_in[7];

    float* out       = (float*)d_out;
    float* o_newxyz  = out;                                   // 8*1024*3
    float* o_feat    = out + (size_t)NB * NPOINT * 3;         // 8*256*1024
    int*   o_inds    = (int*)(o_feat + (size_t)NB * 256 * NPOINT);

    char*   ws      = (char*)d_ws;
    int*    ballbuf = (int*)ws;                               // 8192*32 ints (1 MB)
    half_t* pW1     = (half_t*)(ws + (size_t)NB * NPOINT * NSAMPLE * sizeof(int));
    half_t* pW2     = pW1 + 3 * 4 * 512;                      // 6144 halves
    half_t* pW3     = pW2 + 2 * 8 * 512;                      // 8192 halves
    (void)in_sizes; (void)n_in; (void)out_size; (void)ws_size;

    fps_kernel  <<<NB, 1024, 0, stream>>>(xyz, o_newxyz, o_inds);
    ballq_kernel<<<NB * NPOINT / 8, 256, 0, stream>>>(xyz, o_newxyz, ballbuf);

    pack_w<<<(3 * 4 * 512 + 255) / 256, 256, 0, stream>>>(W1, pW1, 67, 4, 3 * 4 * 512);
    pack_w<<<(2 * 8 * 512 + 255) / 256, 256, 0, stream>>>(W2, pW2, 64, 8, 2 * 8 * 512);
    pack_w<<<(4 * 16 * 512 + 255) / 256, 256, 0, stream>>>(W3, pW3, 128, 16, 4 * 16 * 512);

    mlp_kernel<<<NB * NPOINT / WPB, 128, 0, stream>>>(
        xyz, feat, b1, b2, b3, pW1, pW2, pW3, ballbuf, o_newxyz, o_feat);
}
